// LSTM_17420387352919
// MI455X (gfx1250) — compile-verified, matrix operand order checked by reading, not run on hardware
//
#include <hip/hip_runtime.h>
#include <hip/hip_bf16.h>

// ---------------- problem constants ----------------
#define SD   2
#define H    1024
#define IN_  64
#define B    128
#define S    256
#define KK   1090          // SD + IN_ + H
#define KP   1120          // padded to 35*32 for 16x16x32 bf16 WMMA
#define ZW   (SD + H)      // 1026
#define NWG  64            // 64 hidden blocks of 16  (64*16 = 1024 = H)
#define WGS  256           // 8 waves of 32

// output layout (floats): outputs | z_f | c_z_f | coefficients
#define OUT_ELEMS ((size_t)B * S * ZW)
#define OFF_ZF    (OUT_ELEMS)
#define OFF_CZF   (OFF_ZF + (size_t)B * ZW)
#define OFF_CO    (OFF_CZF + (size_t)B * ZW)

typedef __attribute__((ext_vector_type(16))) __bf16 v16bf;
typedef __attribute__((ext_vector_type(8)))  float  v8f;

union Frag16 {
    v16bf          v;
    uint4          q[2];
    unsigned short us[16];
};

__device__ __forceinline__ unsigned short f2bf(float f) {
    unsigned u = __float_as_uint(f);
    unsigned r = u + 0x7FFFu + ((u >> 16) & 1u);   // round-to-nearest-even
    return (unsigned short)(r >> 16);
}

__device__ __forceinline__ float sigm(float x) { return 1.0f / (1.0f + expf(-x)); }

// --------------- grid-wide barrier (persistent kernel, 64 resident WGs) ---------------
__device__ __forceinline__ void grid_sync(unsigned* cnt, unsigned* gen) {
    __syncthreads();
    if (threadIdx.x == 0) {
        __threadfence();
        unsigned g = __hip_atomic_load(gen, __ATOMIC_ACQUIRE, __HIP_MEMORY_SCOPE_AGENT);
        unsigned a = __hip_atomic_fetch_add(cnt, 1u, __ATOMIC_ACQ_REL, __HIP_MEMORY_SCOPE_AGENT);
        if (a == NWG - 1) {
            __hip_atomic_store(cnt, 0u, __ATOMIC_RELAXED, __HIP_MEMORY_SCOPE_AGENT);
            __hip_atomic_fetch_add(gen, 1u, __ATOMIC_RELEASE, __HIP_MEMORY_SCOPE_AGENT);
        } else {
            while (__hip_atomic_load(gen, __ATOMIC_ACQUIRE, __HIP_MEMORY_SCOPE_AGENT) == g) {
                __builtin_amdgcn_s_sleep(2);
            }
        }
        __threadfence();
    }
    __syncthreads();
}

// --------------- init kernels ---------------
__global__ void k_init_w(const float* __restrict__ WU_w, unsigned short* __restrict__ Wbf) {
    size_t i = (size_t)blockIdx.x * blockDim.x + threadIdx.x;
    if (i >= (size_t)4 * H * KP) return;
    size_t j = i / KP;
    int    k = (int)(i % KP);
    Wbf[i] = (k < KK) ? f2bf(WU_w[j * KK + k]) : (unsigned short)0;
}

__global__ void k_init_state(const float* __restrict__ rnn_input,
                             const float* __restrict__ z0,
                             const float* __restrict__ c0,
                             unsigned short* __restrict__ buf,
                             float* __restrict__ c_state,
                             float* __restrict__ x_state,
                             float* __restrict__ xacc,
                             unsigned* __restrict__ barcnt,
                             unsigned* __restrict__ bargen,
                             float* __restrict__ out) {
    int i = blockIdx.x * blockDim.x + threadIdx.x;
    if (i < 2 * B * KP) {                       // activation double buffer (bf16)
        int par = i / (B * KP);
        int b   = (i / KP) % B;
        int k   = i % KP;
        unsigned short v = 0;
        if (par == 0) {
            if (k >= SD && k < SD + IN_)
                v = f2bf(rnn_input[(size_t)b * S * IN_ + (k - SD)]);       // u_0
            else if (k >= SD + IN_ && k < KK)
                v = f2bf(z0[(size_t)b * ZW + SD + (k - SD - IN_)]);        // h_0
        }
        buf[i] = v;
    }
    if (i < B * H) c_state[i] = c0[(size_t)(i / H) * ZW + SD + (i % H)];
    if (i < 2 * B * SD) {
        int rem = i % (B * SD);
        x_state[i] = z0[(size_t)(rem / SD) * ZW + (rem % SD)];
    }
    if (i < 2 * B * 4) xacc[i] = 0.0f;
    if (i == 0) { *barcnt = 0u; *bargen = 0u; }
    if (i < B * SD) {                            // c_z_f x-part = c0 x-part (never changes)
        int b = i / SD, d = i % SD;
        out[OFF_CZF + (size_t)b * ZW + d] = c0[(size_t)b * ZW + d];
    }
}

// --------------- persistent LTC-LSTM kernel ---------------
__global__ __launch_bounds__(WGS, 1) void ltc_persist(
    const float* __restrict__ rnn_input, const float* __restrict__ tau,
    const float* __restrict__ Amat, const float* __restrict__ WU_b,
    const float* __restrict__ Wa_w, const float* __restrict__ Wa_b,
    const float* __restrict__ Wx_w, const float* __restrict__ Wx_b,
    const unsigned short* __restrict__ Wbf,
    unsigned short* __restrict__ buf,
    float* __restrict__ c_state, float* __restrict__ x_state,
    float* __restrict__ xacc, unsigned* __restrict__ barcnt, unsigned* __restrict__ bargen,
    float* __restrict__ out)
{
    extern __shared__ __align__(16) unsigned short ldsW[];   // [4 gates][16 cols][KP] bf16 = 143,360 B

    const int tid   = threadIdx.x;
    const int wg    = blockIdx.x;        // hidden block
    const int n0    = wg * 16;
    const int mt    = tid >> 5;          // wave index == batch tile
    const int lane  = tid & 31;
    const int lhalf = lane >> 4;
    const int lcol  = lane & 15;
    const int hid   = n0 + lcol;
    const int bx    = mt * 16 + lcol;    // batch row owned by this lane (A rows / x-patch)

    // ---- stage this WG's weight slice into LDS once (reused for all 256 steps) ----
    {
        const int nq = 4 * 16 * (KP / 8);                 // uint4 count
        for (int i = tid; i < nq; i += WGS) {
            int col = i / (KP / 8);                       // gate*16 + col
            int kq  = i % (KP / 8);
            int g   = col >> 4, c = col & 15;
            uint4 v = *(const uint4*)(Wbf + (size_t)(g * H + n0 + c) * KP + (size_t)kq * 8);
            *(uint4*)(ldsW + (size_t)col * KP + (size_t)kq * 8) = v;
        }
    }
    __syncthreads();

    // ---- per-lane constants, loop-invariant ----
    const float bias_i = WU_b[0 * H + hid];
    const float bias_f = WU_b[1 * H + hid];
    const float bias_g = WU_b[2 * H + hid];
    const float bias_o = WU_b[3 * H + hid];
    const float wa0 = Wa_w[0 * H + hid], wa1 = Wa_w[1 * H + hid];
    const float wx0 = Wx_w[0 * H + hid], wx1 = Wx_w[1 * H + hid];
    const float wab0 = Wa_b[0], wab1 = Wa_b[1], wxb0 = Wx_b[0], wxb1 = Wx_b[1];
    const float a00 = Amat[0], a01 = Amat[1], a10 = Amat[2], a11 = Amat[3];

    for (int t = 0; t < S; ++t) {
        const int p = t & 1, q = p ^ 1;
        const unsigned short* bufp = buf + (size_t)p * B * KP;
        unsigned short*       bufq = buf + (size_t)q * B * KP;
        const float* xaccp = xacc + p * (B * 4);
        float*       xaccq = xacc + q * (B * 4);

        // ---- phase A.1: materialize x_{t-1} for this lane's batch row ----
        float px0, px1;
        if (t == 0) {
            px0 = x_state[p * (B * SD) + bx * SD + 0];
            px1 = x_state[p * (B * SD) + bx * SD + 1];
        } else {
            float xp0 = x_state[p * (B * SD) + bx * SD + 0];
            float xp1 = x_state[p * (B * SD) + bx * SD + 1];
            float al0 = sigm(xaccp[bx * 4 + 0] + wab0);
            float al1 = sigm(xaccp[bx * 4 + 1] + wab1);
            float xt0 = xaccp[bx * 4 + 2] + wxb0;
            float xt1 = xaccp[bx * 4 + 3] + wxb1;
            float tp  = tau[(size_t)bx * S + (t - 1)];
            float xm0 = xp0 + tp * (a00 * xp0 + a01 * xp1);
            float xm1 = xp1 + tp * (a10 * xp0 + a11 * xp1);
            px0 = al0 * xm0 + (1.0f - al0) * xt0;
            px1 = al1 * xm1 + (1.0f - al1) * xt1;
            if (wg == 0 && lhalf == 0) {            // WG0 owns the x/alpha side outputs
                out[(size_t)bx * S * ZW + (size_t)(t - 1) * ZW + 0] = px0;
                out[(size_t)bx * S * ZW + (size_t)(t - 1) * ZW + 1] = px1;
                out[OFF_CO + (size_t)bx * S * SD + (size_t)(t - 1) * SD + 0] = al0;
                out[OFF_CO + (size_t)bx * S * SD + (size_t)(t - 1) * SD + 1] = al1;
                x_state[q * (B * SD) + bx * SD + 0] = px0;
                x_state[q * (B * SD) + bx * SD + 1] = px1;
            }
        }

        // ---- phase A.2: gates GEMM (bf16 WMMA, fp32 accum), B-matrix from LDS ----
        v8f acc0 = {}, acc1 = {}, acc2 = {}, acc3 = {};
        const unsigned short* arow = bufp + (size_t)bx * KP + (lhalf ? 8 : 0);
        for (int kb = 0; kb < KP; kb += 32) {
            Frag16 fa;
            fa.q[0] = *(const uint4*)(arow + kb);
            fa.q[1] = *(const uint4*)(arow + kb + 16);
            if (kb == 0 && lhalf == 0) {            // patch x_{t-1} into K=0,1
                fa.us[0] = f2bf(px0);
                fa.us[1] = f2bf(px1);
            }
            const unsigned short* bb = ldsW + (size_t)lcol * KP + kb + (lhalf ? 8 : 0);
            Frag16 fb0, fb1, fb2, fb3;
            fb0.q[0] = *(const uint4*)(bb + 0 * 16 * KP);  fb0.q[1] = *(const uint4*)(bb + 0 * 16 * KP + 16);
            fb1.q[0] = *(const uint4*)(bb + 1 * 16 * KP);  fb1.q[1] = *(const uint4*)(bb + 1 * 16 * KP + 16);
            fb2.q[0] = *(const uint4*)(bb + 2 * 16 * KP);  fb2.q[1] = *(const uint4*)(bb + 2 * 16 * KP + 16);
            fb3.q[0] = *(const uint4*)(bb + 3 * 16 * KP);  fb3.q[1] = *(const uint4*)(bb + 3 * 16 * KP + 16);
            acc0 = __builtin_amdgcn_wmma_f32_16x16x32_bf16(false, fa.v, false, fb0.v, (short)0, acc0, false, false);
            acc1 = __builtin_amdgcn_wmma_f32_16x16x32_bf16(false, fa.v, false, fb1.v, (short)0, acc1, false, false);
            acc2 = __builtin_amdgcn_wmma_f32_16x16x32_bf16(false, fa.v, false, fb2.v, (short)0, acc2, false, false);
            acc3 = __builtin_amdgcn_wmma_f32_16x16x32_bf16(false, fa.v, false, fb3.v, (short)0, acc3, false, false);
        }

        // ---- phase A.3: LSTM cell + h side-effects + Wa/Wx partial dots ----
        for (int r = 0; r < 8; ++r) {
            int b = mt * 16 + r + 8 * lhalf;        // C/D layout: M = r + 8*(lane>=16)
            float iv = acc0[r] + bias_i;
            float fv = acc1[r] + bias_f;
            float gv = acc2[r] + bias_g;
            float ov = acc3[r] + bias_o;
            size_t ci = (size_t)b * H + hid;
            float c_old = c_state[ci];
            float cn = sigm(fv) * c_old + sigm(iv) * tanhf(gv);
            float hn = sigm(ov) * tanhf(cn);
            c_state[ci] = cn;
            out[(size_t)b * S * ZW + (size_t)t * ZW + SD + hid] = hn;
            bufq[(size_t)b * KP + SD + IN_ + hid] = f2bf(hn);
            if (t == S - 1) {
                out[OFF_ZF  + (size_t)b * ZW + SD + hid] = hn;
                out[OFF_CZF + (size_t)b * ZW + SD + hid] = cn;
            }
            // partial dot products for alpha / x_tilde, reduce over the 16 hid lanes
            float s0 = hn * wa0, s1 = hn * wa1, s2 = hn * wx0, s3 = hn * wx1;
            for (int off = 8; off >= 1; off >>= 1) {
                s0 += __shfl_xor(s0, off, 32);
                s1 += __shfl_xor(s1, off, 32);
                s2 += __shfl_xor(s2, off, 32);
                s3 += __shfl_xor(s3, off, 32);
            }
            if (lcol == 0) {
                atomicAdd(&xaccq[b * 4 + 0], s0);
                atomicAdd(&xaccq[b * 4 + 1], s1);
                atomicAdd(&xaccq[b * 4 + 2], s2);
                atomicAdd(&xaccq[b * 4 + 3], s3);
            }
        }

        // ---- phase A.4: WG1 stages u_{t+1} into the next-parity buffer ----
        if (wg == 1 && (t + 1) < S) {
            for (int i = tid; i < B * IN_; i += WGS) {
                int b = i >> 6, c = i & 63;
                bufq[(size_t)b * KP + SD + c] =
                    f2bf(rnn_input[(size_t)b * S * IN_ + (size_t)(t + 1) * IN_ + c]);
            }
        }

        grid_sync(barcnt, bargen);
        // ---- phase B: zero xacc[p] for reuse as the accumulator of step t+1 ----
        if (wg == 0) {
            for (int i = tid; i < B * 4; i += WGS) xacc[p * (B * 4) + i] = 0.0f;
        }
        grid_sync(barcnt, bargen);
    }

    // ---- tail: x_{S-1}, coefficients[S-1], z_f x-part ----
    if (wg == 0 && lhalf == 0) {
        const int p = S & 1;                        // 0
        const float* xaccp = xacc + p * (B * 4);
        float xp0 = x_state[p * (B * SD) + bx * SD + 0];
        float xp1 = x_state[p * (B * SD) + bx * SD + 1];
        float al0 = sigm(xaccp[bx * 4 + 0] + wab0);
        float al1 = sigm(xaccp[bx * 4 + 1] + wab1);
        float xt0 = xaccp[bx * 4 + 2] + wxb0;
        float xt1 = xaccp[bx * 4 + 3] + wxb1;
        float tp  = tau[(size_t)bx * S + (S - 1)];
        float xm0 = xp0 + tp * (a00 * xp0 + a01 * xp1);
        float xm1 = xp1 + tp * (a10 * xp0 + a11 * xp1);
        float px0 = al0 * xm0 + (1.0f - al0) * xt0;
        float px1 = al1 * xm1 + (1.0f - al1) * xt1;
        out[(size_t)bx * S * ZW + (size_t)(S - 1) * ZW + 0] = px0;
        out[(size_t)bx * S * ZW + (size_t)(S - 1) * ZW + 1] = px1;
        out[OFF_CO + (size_t)bx * S * SD + (size_t)(S - 1) * SD + 0] = al0;
        out[OFF_CO + (size_t)bx * S * SD + (size_t)(S - 1) * SD + 1] = al1;
        out[OFF_ZF + (size_t)bx * ZW + 0] = px0;
        out[OFF_ZF + (size_t)bx * ZW + 1] = px1;
    }
}

// --------------- host-side launcher ---------------
extern "C" void kernel_launch(void* const* d_in, const int* in_sizes, int n_in,
                              void* d_out, int out_size, void* d_ws, size_t ws_size,
                              hipStream_t stream) {
    const float* rnn_input = (const float*)d_in[0];
    const float* z0        = (const float*)d_in[1];
    const float* c0        = (const float*)d_in[2];
    const float* tau       = (const float*)d_in[3];
    const float* Amat      = (const float*)d_in[4];
    const float* WU_w      = (const float*)d_in[5];
    const float* WU_b      = (const float*)d_in[6];
    const float* Wa_w      = (const float*)d_in[7];
    const float* Wa_b      = (const float*)d_in[8];
    const float* Wx_w      = (const float*)d_in[9];
    const float* Wx_b      = (const float*)d_in[10];
    float* out = (float*)d_out;

    char* ws = (char*)d_ws;
    size_t off = 0;
    auto alloc = [&](size_t bytes) -> void* {
        off = (off + 255) & ~(size_t)255;
        void* pp = ws + off;
        off += bytes;
        return pp;
    };
    unsigned short* Wbf     = (unsigned short*)alloc((size_t)4 * H * KP * 2);
    unsigned short* buf     = (unsigned short*)alloc((size_t)2 * B * KP * 2);
    float*          c_state = (float*)alloc((size_t)B * H * 4);
    float*          x_state = (float*)alloc((size_t)2 * B * SD * 4);
    float*          xacc    = (float*)alloc((size_t)2 * B * 4 * 4);
    unsigned*       barcnt  = (unsigned*)alloc(128);
    unsigned*       bargen  = (unsigned*)alloc(128);

    {
        size_t n = (size_t)4 * H * KP;
        k_init_w<<<(unsigned)((n + 255) / 256), 256, 0, stream>>>(WU_w, Wbf);
    }
    {
        int n = 2 * B * KP;   // dominates all other init ranges
        k_init_state<<<(n + 255) / 256, 256, 0, stream>>>(rnn_input, z0, c0, buf, c_state,
                                                          x_state, xacc, barcnt, bargen, out);
    }
    size_t ldsBytes = (size_t)4 * 16 * KP * sizeof(unsigned short);   // 143,360 B < 320 KB
    ltc_persist<<<NWG, WGS, ldsBytes, stream>>>(rnn_input, tau, Amat, WU_b, Wa_w, Wa_b,
                                                Wx_w, Wx_b, Wbf, buf, c_state, x_state,
                                                xacc, barcnt, bargen, out);
}
